// GraphAttentionLayer_52158082843308
// MI455X (gfx1250) — compile-verified, matrix operand order checked by reading, not run on hardware
//
#include <hip/hip_runtime.h>
#include <cstdint>

#define N_NODES 8192
#define F_IN    512
#define F_OUT   64
#define ALPHA   0.2f
#define JPARTS  8
#define JSPAN   (N_NODES / JPARTS)   // 1024
#define CHUNK   32
#define NCHUNK  (JSPAN / CHUNK)      // 32

typedef __attribute__((ext_vector_type(16))) __bf16 v16bf;
typedef __attribute__((ext_vector_type(8)))  float  v8f;

struct U8 { unsigned int v[8]; };   // 32 bytes == v16bf

static __device__ __forceinline__ unsigned int f2bf1(float f) {
  unsigned int u = __float_as_uint(f);
  return (u + 0x7FFFu + ((u >> 16) & 1u)) >> 16;   // RNE
}
static __device__ __forceinline__ unsigned int pack2(float a, float b) {
  return f2bf1(a) | (f2bf1(b) << 16);
}
// split f32 pair into bf16-hi pack and exact-residual bf16-lo pack
static __device__ __forceinline__ void pack_hilo(float a, float b,
                                                 unsigned int& hi, unsigned int& lo) {
  const unsigned int ha = f2bf1(a), hb = f2bf1(b);
  hi = ha | (hb << 16);
  const float ra = a - __uint_as_float(ha << 16);   // exact
  const float rb = b - __uint_as_float(hb << 16);   // exact
  lo = pack2(ra, rb);
}
static __device__ __forceinline__ int fOrd(float x) {
  int b = __float_as_int(x);
  return b >= 0 ? b : (b ^ 0x7fffffff);
}
static __device__ __forceinline__ float ordToF(int o) {
  int b = o >= 0 ? o : (o ^ 0x7fffffff);
  return __int_as_float(b);
}
static __device__ __forceinline__ v16bf bc16(const U8& u) {
  return __builtin_bit_cast(v16bf, u);
}
static __device__ __forceinline__ v8f wmma_bf16(v16bf a, v16bf b, v8f c) {
  return __builtin_amdgcn_wmma_f32_16x16x32_bf16(false, a, false, b,
                                                 (short)0, c, false, false);
}

// ---------------------------------------------------------------- init
__global__ void gat_init(int* e2ord) {
  if (threadIdx.x == 0 && blockIdx.x == 0) *e2ord = INT_MIN;
}

// ---------------------------------------------------------------- Wh = h @ W
// 1 wave -> 16 rows x 64 cols. split-bf16 (hi+lo) WMMA, ~fp32 accurate.
__global__ void gat_wh_gemm(const float* __restrict__ h,
                            const float* __restrict__ W,
                            float* __restrict__ Whf,            // [N][64] fp32
                            unsigned short* __restrict__ WhThi, // [64][N] bf16 hi
                            unsigned short* __restrict__ WhTlo) // [64][N] bf16 lo
{
  const int lane = threadIdx.x & 31;
  const int wave = threadIdx.x >> 5;
  const int tile = blockIdx.x * (blockDim.x >> 5) + wave;   // 0..511
  const int rowbase = tile * 16;
  const int M  = lane & 15;
  const int kb = (lane < 16) ? 0 : 8;
  const int row = rowbase + M;

  v8f c[4] = {};

  for (int kc = 0; kc < F_IN; kc += 32) {
    // ---- A tile (h rows) in WMMA A lane layout, hi+lo split
    const float* ap = h + (size_t)row * F_IN + kc + kb;
    float4 f0 = *(const float4*)(ap + 0);
    float4 f1 = *(const float4*)(ap + 4);
    float4 f2 = *(const float4*)(ap + 16);
    float4 f3 = *(const float4*)(ap + 20);
    U8 ahi, alo;
    pack_hilo(f0.x, f0.y, ahi.v[0], alo.v[0]);
    pack_hilo(f0.z, f0.w, ahi.v[1], alo.v[1]);
    pack_hilo(f1.x, f1.y, ahi.v[2], alo.v[2]);
    pack_hilo(f1.z, f1.w, ahi.v[3], alo.v[3]);
    pack_hilo(f2.x, f2.y, ahi.v[4], alo.v[4]);
    pack_hilo(f2.z, f2.w, ahi.v[5], alo.v[5]);
    pack_hilo(f3.x, f3.y, ahi.v[6], alo.v[6]);
    pack_hilo(f3.z, f3.w, ahi.v[7], alo.v[7]);
    const v16bf Ahi = bc16(ahi), Alo = bc16(alo);

    // ---- 4 B tiles of W (K x 16), lane col = t*16 + M, hi+lo split
#pragma unroll
    for (int t = 0; t < 4; ++t) {
      const int n = t * 16 + M;
      U8 bhi, blo;
#pragma unroll
      for (int i = 0; i < 4; ++i) {
        const int k0 = kc + kb + 2 * i;
        const int k1 = kc + 16 + kb + 2 * i;
        pack_hilo(W[(size_t)k0 * F_OUT + n], W[(size_t)(k0 + 1) * F_OUT + n],
                  bhi.v[i], blo.v[i]);
        pack_hilo(W[(size_t)k1 * F_OUT + n], W[(size_t)(k1 + 1) * F_OUT + n],
                  bhi.v[4 + i], blo.v[4 + i]);
      }
      const v16bf Bhi = bc16(bhi), Blo = bc16(blo);
      c[t] = wmma_bf16(Ahi, Bhi, c[t]);   // hh
      c[t] = wmma_bf16(Alo, Bhi, c[t]);   // lh
      c[t] = wmma_bf16(Ahi, Blo, c[t]);   // hl
    }
  }

  // ---- store D tiles: lane holds N=L&15, rows r + (L>=16 ? 8 : 0)
  const int moff = (lane >= 16) ? 8 : 0;
#pragma unroll
  for (int t = 0; t < 4; ++t) {
    const int col = t * 16 + M;
#pragma unroll
    for (int r = 0; r < 8; ++r) {
      const int gr = rowbase + r + moff;
      const float v = c[t][r];
      Whf[(size_t)gr * F_OUT + col] = v;
      const unsigned int hb = f2bf1(v);
      WhThi[(size_t)col * N_NODES + gr] = (unsigned short)hb;
      WhTlo[(size_t)col * N_NODES + gr] =
          (unsigned short)f2bf1(v - __uint_as_float(hb << 16));
    }
  }
}

// ---------------------------------------------------------------- e1,e2 + max(e2)
__global__ void gat_evec(const float* __restrict__ Whf,
                         const float* __restrict__ a,
                         float* __restrict__ e1, float* __restrict__ e2,
                         int* __restrict__ e2ord)
{
  const int i = blockIdx.x * blockDim.x + threadIdx.x;  // 0..8191
  float s1 = 0.f, s2 = 0.f;
  const float4* wp = (const float4*)(Whf + (size_t)i * F_OUT);
  const float4* a1 = (const float4*)(a);
  const float4* a2 = (const float4*)(a + F_OUT);
#pragma unroll
  for (int q = 0; q < F_OUT / 4; ++q) {
    float4 w = wp[q], u = a1[q], v = a2[q];
    s1 += w.x * u.x + w.y * u.y + w.z * u.z + w.w * u.w;
    s2 += w.x * v.x + w.y * v.y + w.z * v.z + w.w * v.w;
  }
  e1[i] = s1;
  e2[i] = s2;

  __shared__ float red[256];
  red[threadIdx.x] = s2;
  __syncthreads();
  for (int off = 128; off > 0; off >>= 1) {
    if ((int)threadIdx.x < off)
      red[threadIdx.x] = fmaxf(red[threadIdx.x], red[threadIdx.x + off]);
    __syncthreads();
  }
  if (threadIdx.x == 0) atomicMax(e2ord, fOrd(red[0]));
}

// ---------------------------------------------------------------- fused attention
// block = 8 waves sharing one j-part; Wh chunk staged in LDS (double-buffered).
// Fixed per-row upper-bound softmax shift => partials over j combine linearly.
static __device__ __forceinline__ float pcalc(int adjv, float e2v, float e1v, float m) {
  const float t = e1v + e2v;
  const float e = t > 0.f ? t : ALPHA * t;
  return adjv > 0 ? __expf(e - m) : 0.f;
}

#define LROW 20   // 16 data dwords + 4 pad (keeps 16B alignment, spreads banks)

__global__ void gat_attn(const int* __restrict__ adj,
                         const float* __restrict__ e1,
                         const float* __restrict__ e2,
                         const unsigned short* __restrict__ WhThi,
                         const unsigned short* __restrict__ WhTlo,
                         const int* __restrict__ e2ord,
                         float* __restrict__ accPart,   // [JPARTS][N][64]
                         float* __restrict__ sPart)     // [JPARTS][N]
{
  __shared__ unsigned int lds[2][2][64][LROW];   // [buf][hi/lo][n][dwords] = 20KB

  const int lane = threadIdx.x & 31;
  const int wave = threadIdx.x >> 5;                   // 0..7
  const int part = blockIdx.x & (JPARTS - 1);
  const int tg   = blockIdx.x >> 3;                    // 0..63
  const int tile = tg * 8 + wave;                      // 0..511
  const int rowbase = tile * 16;
  const int M  = lane & 15;
  const int kb = (lane < 16) ? 0 : 8;
  const int kb4 = (lane < 16) ? 0 : 1;                 // uint4 index offset
  const int row = rowbase + M;
  const int jstart = part * JSPAN;

  // cooperative-fill mapping: 256 threads x 16B (hi) + 16B (lo) per chunk
  const int fn = threadIdx.x >> 2;      // 0..63  (column n)
  const int fs = threadIdx.x & 3;       // 0..3   (8-short segment)

  const float e2max = ordToF(*e2ord);
  const float e1v = e1[row];
  const float t0  = e1v + e2max;
  const float m   = t0 > 0.f ? t0 : ALPHA * t0;

  v8f c[4] = {};
  float srow = 0.f;
  const int* adjp = adj + (size_t)row * N_NODES;

  // prologue: fill buffer 0 with chunk 0
  {
    const size_t src = (size_t)fn * N_NODES + jstart + fs * 8;
    *(uint4*)&lds[0][0][fn][fs * 4] = *(const uint4*)(WhThi + src);
    *(uint4*)&lds[0][1][fn][fs * 4] = *(const uint4*)(WhTlo + src);
  }

  for (int ci = 0; ci < NCHUNK; ++ci) {
    const int jb = jstart + ci * CHUNK;
    const int buf = ci & 1;
    __syncthreads();                       // fill(ci) done; reads of buf^1 done

    if (ci + 1 < NCHUNK) {                 // prefetch next chunk into other buffer
      const size_t src = (size_t)fn * N_NODES + (jb + CHUNK) + fs * 8;
      *(uint4*)&lds[buf ^ 1][0][fn][fs * 4] = *(const uint4*)(WhThi + src);
      *(uint4*)&lds[buf ^ 1][1][fn][fs * 4] = *(const uint4*)(WhTlo + src);
    }
    __builtin_prefetch(adjp + jb + 256, 0, 1);   // global_prefetch_b8, 1KB ahead

    // ---- adjacency + e2 for this lane's 16 j's (WMMA A lane layout)
    const int4 a0 = *(const int4*)(adjp + jb + kb);
    const int4 a1 = *(const int4*)(adjp + jb + kb + 4);
    const int4 a2 = *(const int4*)(adjp + jb + 16 + kb);
    const int4 a3 = *(const int4*)(adjp + jb + 16 + kb + 4);
    const float4 q0 = *(const float4*)(e2 + jb + kb);
    const float4 q1 = *(const float4*)(e2 + jb + kb + 4);
    const float4 q2 = *(const float4*)(e2 + jb + 16 + kb);
    const float4 q3 = *(const float4*)(e2 + jb + 16 + kb + 4);

    float pv[16];
    pv[0]  = pcalc(a0.x, q0.x, e1v, m); pv[1]  = pcalc(a0.y, q0.y, e1v, m);
    pv[2]  = pcalc(a0.z, q0.z, e1v, m); pv[3]  = pcalc(a0.w, q0.w, e1v, m);
    pv[4]  = pcalc(a1.x, q1.x, e1v, m); pv[5]  = pcalc(a1.y, q1.y, e1v, m);
    pv[6]  = pcalc(a1.z, q1.z, e1v, m); pv[7]  = pcalc(a1.w, q1.w, e1v, m);
    pv[8]  = pcalc(a2.x, q2.x, e1v, m); pv[9]  = pcalc(a2.y, q2.y, e1v, m);
    pv[10] = pcalc(a2.z, q2.z, e1v, m); pv[11] = pcalc(a2.w, q2.w, e1v, m);
    pv[12] = pcalc(a3.x, q3.x, e1v, m); pv[13] = pcalc(a3.y, q3.y, e1v, m);
    pv[14] = pcalc(a3.z, q3.z, e1v, m); pv[15] = pcalc(a3.w, q3.w, e1v, m);

#pragma unroll
    for (int q = 0; q < 16; ++q) srow += pv[q];

    U8 ahi, alo;
#pragma unroll
    for (int q = 0; q < 8; ++q)
      pack_hilo(pv[2 * q], pv[2 * q + 1], ahi.v[q], alo.v[q]);
    const v16bf Ahi = bc16(ahi), Alo = bc16(alo);

    // ---- 4 n-tiles, B from LDS, 3-term split-bf16 accumulate
#pragma unroll
    for (int t = 0; t < 4; ++t) {
      const uint4* rh = (const uint4*)&lds[buf][0][t * 16 + M][0];
      const uint4* rl = (const uint4*)&lds[buf][1][t * 16 + M][0];
      const uint4 h0 = rh[kb4],     h1 = rh[2 + kb4];
      const uint4 l0 = rl[kb4],     l1 = rl[2 + kb4];
      U8 bhi, blo;
      bhi.v[0] = h0.x; bhi.v[1] = h0.y; bhi.v[2] = h0.z; bhi.v[3] = h0.w;
      bhi.v[4] = h1.x; bhi.v[5] = h1.y; bhi.v[6] = h1.z; bhi.v[7] = h1.w;
      blo.v[0] = l0.x; blo.v[1] = l0.y; blo.v[2] = l0.z; blo.v[3] = l0.w;
      blo.v[4] = l1.x; blo.v[5] = l1.y; blo.v[6] = l1.z; blo.v[7] = l1.w;
      const v16bf Bhi = bc16(bhi), Blo = bc16(blo);
      c[t] = wmma_bf16(Ahi, Bhi, c[t]);   // hh
      c[t] = wmma_bf16(Alo, Bhi, c[t]);   // lh
      c[t] = wmma_bf16(Ahi, Blo, c[t]);   // hl
    }
  }

  // per-row exp-sum: combine lane L with L^16 (wave32)
  const float s = srow + __shfl_xor(srow, 16, 32);
  if (lane < 16) sPart[part * N_NODES + row] = s;

  float* op = accPart + (size_t)part * N_NODES * F_OUT;
  const int moff = (lane >= 16) ? 8 : 0;
#pragma unroll
  for (int t = 0; t < 4; ++t) {
    const int col = t * 16 + M;
#pragma unroll
    for (int r = 0; r < 8; ++r)
      op[(size_t)(rowbase + r + moff) * F_OUT + col] = c[t][r];
  }
}

// ---------------------------------------------------------------- combine + ELU
__global__ void gat_combine(const float* __restrict__ accPart,
                            const float* __restrict__ sPart,
                            float* __restrict__ out)
{
  const int idx = blockIdx.x * blockDim.x + threadIdx.x; // 0 .. N*64-1
  const int i = idx >> 6;
  float s = 0.f, v = 0.f;
#pragma unroll
  for (int p = 0; p < JPARTS; ++p) {
    s += sPart[p * N_NODES + i];
    v += accPart[(size_t)p * N_NODES * F_OUT + idx];
  }
  const float hp = v / fmaxf(s, 1e-30f);
  out[idx] = hp > 0.f ? hp : (__expf(hp) - 1.f);   // ELU, alpha=1
}

// ---------------------------------------------------------------- launch
extern "C" void kernel_launch(void* const* d_in, const int* in_sizes, int n_in,
                              void* d_out, int out_size, void* d_ws, size_t ws_size,
                              hipStream_t stream) {
  const float* h   = (const float*)d_in[0];
  const int*   adj = (const int*)d_in[1];
  const float* W   = (const float*)d_in[2];
  const float* a   = (const float*)d_in[3];
  float* out = (float*)d_out;

  // workspace carve (256-B aligned), ~20.5 MB total
  uintptr_t base = (uintptr_t)d_ws;
  auto take = [&](size_t bytes) {
    uintptr_t p = base;
    base += (bytes + 255) & ~(size_t)255;
    return (void*)p;
  };
  float*          Whf     = (float*)take((size_t)N_NODES * F_OUT * 4);
  unsigned short* WhThi   = (unsigned short*)take((size_t)F_OUT * N_NODES * 2);
  unsigned short* WhTlo   = (unsigned short*)take((size_t)F_OUT * N_NODES * 2);
  float*          e1      = (float*)take((size_t)N_NODES * 4);
  float*          e2      = (float*)take((size_t)N_NODES * 4);
  int*            e2ord   = (int*)take(256);
  float*          sPart   = (float*)take((size_t)JPARTS * N_NODES * 4);
  float*          accPart = (float*)take((size_t)JPARTS * N_NODES * F_OUT * 4);

  gat_init<<<1, 32, 0, stream>>>(e2ord);
  gat_wh_gemm<<<128, 128, 0, stream>>>(h, W, Whf, WhThi, WhTlo);   // 512 tiles
  gat_evec<<<N_NODES / 256, 256, 0, stream>>>(Whf, a, e1, e2, e2ord);
  gat_attn<<<64 * JPARTS, 256, 0, stream>>>(adj, e1, e2, WhThi, WhTlo, e2ord,
                                            accPart, sPart);
  gat_combine<<<(N_NODES * F_OUT) / 256, 256, 0, stream>>>(accPart, sPart, out);
}
